// SimpleLSTM_74165495267698
// MI455X (gfx1250) — matrix operand order repaired and checked by reading, not compiled, not run on hardware
//
#include <hip/hip_runtime.h>
#include <hip/hip_bf16.h>
#include <math.h>

// ---------------------------------------------------------------------------
// Persistent-RNN 2-layer LSTM for MI455X (gfx1250, wave32, WMMA).
//
//   Phase A: layer-0 LSTM, fused K=10 input projection, streams h1 (f16) to ws
//            8 waves, wave-local gates: wave w owns n-tiles {g*128+16w}, the
//            i/f/g/o gates of its own cell columns -> update reads gates
//            straight from WMMA accumulators, 2 barriers/step.
//   Phase B: layer-1 LSTM reading h1 tiles from ws + final FC(128->1)
//            16 waves x 2 n-tiles, gates exchanged via LDS.
//
// Recurrent weights live in VGPRs as pre-packed f16 WMMA B-fragments for all
// 512 timesteps; opaque asm barriers pin them (and the bias splats) in
// materialized form so the register allocator cannot rematerialize the
// f32->f16 conversions in-loop. The first WMMA of each gate chain consumes
// the resident bias splat directly as the C operand (D != C), eliminating
// per-iteration accumulator-init moves. Activations are branch-free
// (native v_tanh_f32 + v_exp/v_rcp).
// Workspace requirement: B*T*H*2 = 128 MiB (f16 h1 sequence).
// ---------------------------------------------------------------------------

#define IN_   10
#define H_    128
#define Bsz_  1024
#define T_    512
#define G4H_  512            // 4*H
#define BT_   16             // batch tile == WMMA M
#define NBLK_ (Bsz_ / BT_)   // 64 workgroups
#define NTHR_A_ 256          // phase A: 8 wave32
#define NTHR_B_ 512          // phase B: 16 wave32
#define NGATE_ 4             // phase A n-tiles per wave (one per LSTM gate)
#define NT_PW_ 2             // phase B n-tiles per wave

typedef _Float16 v16h __attribute__((ext_vector_type(16)));
typedef _Float16 v8h  __attribute__((ext_vector_type(8)));
typedef float    v8f  __attribute__((ext_vector_type(8)));
typedef float    v4f  __attribute__((ext_vector_type(4)));

union Frag16 { v16h v; v8h h[2]; };

// Opaque register barriers: force materialization, kill rematerialization.
__device__ __forceinline__ void pin_frag(v16h& f) { asm volatile("" : "+v"(f)); }
__device__ __forceinline__ void pin_acc(v8f& f)   { asm volatile("" : "+v"(f)); }

__device__ __forceinline__ v8f wmma_16x16x32(v16h a, v16h b, v8f c) {
  // D = A(16x32 f16) * B(32x16 f16) + C(16x16 f32)
  return __builtin_amdgcn_wmma_f32_16x16x32_f16(false, a, false, b,
                                                (short)0, c, false, false);
}

// ---- branch-free fast transcendentals -------------------------------------
#define LOG2E_  1.44269504088896340736f

__device__ __forceinline__ float fast_exp2(float x) {
#if __has_builtin(__builtin_amdgcn_exp2f)
  return __builtin_amdgcn_exp2f(x);        // v_exp_f32
#else
  return exp2f(x);
#endif
}

__device__ __forceinline__ float fast_rcp(float x) {
#if __has_builtin(__builtin_amdgcn_rcpf)
  return __builtin_amdgcn_rcpf(x);         // v_rcp_f32
#else
  return 1.0f / x;
#endif
}

__device__ __forceinline__ float fast_sigmoid(float x) {
  // 1/(1+exp(-x)); exp2 under/overflow saturates cleanly to 1 / 0.
  return fast_rcp(1.0f + fast_exp2(-LOG2E_ * x));
}

__device__ __forceinline__ float fast_tanh(float x) {
#if __has_builtin(__builtin_amdgcn_tanhf)
  return __builtin_amdgcn_tanhf(x);        // v_tanh_f32 (confirmed in round 2)
#elif __has_builtin(__builtin_amdgcn_tanh_f32)
  return __builtin_amdgcn_tanh_f32(x);
#else
  const float ax = __builtin_fabsf(x);
  const float e  = fast_exp2(-2.0f * LOG2E_ * ax);
  const float r  = (1.0f - e) * fast_rcp(1.0f + e);
  return __builtin_copysignf(r, x);
#endif
}

// A fragment: 16xK f16 row-major tile (ldk halves per row).
// Lane l (<16): row m=l, K-halves {kb..kb+15}, kb = kt*32.
// Lane l (>=16): row m=l-16, same but kb += 16.
__device__ __forceinline__ v16h load_a_frag(const _Float16* base, int ldk,
                                            int lane, int kt) {
  const int m  = lane & 15;
  const int kb = kt * 32 + ((lane >> 4) & 1) * 16;
  Frag16 f;
  f.h[0] = *(const v8h*)(base + m * ldk + kb);
  f.h[1] = *(const v8h*)(base + m * ldk + kb + 8);
  return f.v;
}

// B fragment from f32 weight matrix W[G4H][ldk] (gate-major, as in PyTorch):
// B = W^T (KxN). Lane holds column n, a contiguous 16-run of K:
// lanes 0-15 -> K = kt*32 + 0..15, lanes 16-31 -> K = kt*32 + 16..31.
__device__ __forceinline__ v16h load_b_frag_w(const float* __restrict__ W,
                                              int ldk, int n, int lane, int kt) {
  const int kb = kt * 32 + ((lane >> 4) & 1) * 16;
  const float* p = W + (size_t)n * ldk + kb;
  v16h r;
#pragma unroll
  for (int i = 0; i < 16; i += 4) {
    v4f q = *(const v4f*)(p + i);
#pragma unroll
    for (int j = 0; j < 4; ++j) r[i + j] = (_Float16)q[j];
  }
  return r;
}

// ---------------------------------------------------------------------------
// Phase A: layer-0 LSTM.  gates = [x_t|pad] @ Wih0^T + h @ Whh0^T + bias
// 8 waves; wave w owns the i/f/g/o gate tiles of cell columns [16w,16w+16).
// ---------------------------------------------------------------------------
__global__ __launch_bounds__(NTHR_A_, 1)
void lstm_layer0_kernel(const float* __restrict__ x,
                        const float* __restrict__ Wih,
                        const float* __restrict__ Whh,
                        const float* __restrict__ bih,
                        const float* __restrict__ bhh,
                        _Float16* __restrict__ h1seq) {
  __shared__ alignas(16) _Float16 xA[BT_][32];     // x_t padded to K=32
  __shared__ alignas(16) _Float16 hA[BT_][H_];     // h state (f16 A-matrix)

  const int tid  = threadIdx.x;
  const int lane = tid & 31;
  const int w    = tid >> 5;
  const int hi   = (lane >> 4) & 1;
  const int b0   = blockIdx.x * BT_;
  const int jcol = w * BT_ + (lane & 15);          // this lane's cell column

  // --- weights + bias splats resident in VGPRs, loaded once -------------
  v16h wx[NGATE_];          // Wih0 (K=32 padded), 1 k-tile per gate
  v16h wh[NGATE_][4];       // Whh0, K=128 -> 4 k-tiles per gate
  v8f  bv[NGATE_];          // bias splat, used as C of the first WMMA
#pragma unroll
  for (int g = 0; g < NGATE_; ++g) {
    const int n = g * H_ + jcol;
    const float b = bih[n] + bhh[n];
#pragma unroll
    for (int r = 0; r < 8; ++r) bv[g][r] = b;
#pragma unroll
    for (int kt = 0; kt < 4; ++kt)
      wh[g][kt] = load_b_frag_w(Whh, H_, n, lane, kt);
    // Wih0 is [512][10]; hi lane-half covers K=16..31 -> all zero pad
    v16h r;
#pragma unroll
    for (int i = 0; i < 16; ++i) r[i] = (_Float16)0.0f;
    if (!hi) {
      const float* p = Wih + (size_t)n * IN_;
#pragma unroll
      for (int i = 0; i < IN_; ++i) r[i] = (_Float16)p[i];
    }
    wx[g] = r;
  }
  // Pin in materialized form (defeat rematerialization).
#pragma unroll
  for (int g = 0; g < NGATE_; ++g) {
    pin_frag(wx[g]);
    pin_acc(bv[g]);
#pragma unroll
    for (int kt = 0; kt < 4; ++kt) pin_frag(wh[g][kt]);
  }

  // --- init state -------------------------------------------------------
  for (int i = tid; i < BT_ * 32; i += NTHR_A_) (&xA[0][0])[i] = (_Float16)0.0f;
  for (int i = tid; i < BT_ * H_; i += NTHR_A_) (&hA[0][0])[i] = (_Float16)0.0f;
  float c[8];
#pragma unroll
  for (int r = 0; r < 8; ++r) c[r] = 0.0f;

#pragma unroll 1
  for (int t = 0; t < T_; ++t) {
    // stage x_t tile (waves 0-4 only; wave-uniform branch)
    if (tid < BT_ * IN_) {
      const int m = tid / IN_;
      const int k = tid - m * IN_;
      const size_t idx = ((size_t)(b0 + m) * T_ + t) * IN_ + k;
      xA[m][k] = (_Float16)x[idx];
      if (t + 1 < T_) __builtin_prefetch(&x[idx + IN_], 0, 0);  // global_prefetch_b8
    }
    __syncthreads();   // h(t-1) + x_t visible

    // stream h_{t-1} to workspace (overlaps with this step's GEMM)
    if (t > 0) {
      const v8h* s = (const v8h*)(&hA[0][0]) + tid;            // 256 * 8 halves
      v8h* d = (v8h*)(h1seq + ((size_t)blockIdx.x * T_ + (t - 1)) * (BT_ * H_)) + tid;
      *d = *s;
    }

    // --- gate GEMM: 20 WMMAs; bias splat enters as C of the 1st WMMA ---
    v8f acc[NGATE_];
    {
      const v16h a = load_a_frag(&xA[0][0], 32, lane, 0);
#pragma unroll
      for (int g = 0; g < NGATE_; ++g) acc[g] = wmma_16x16x32(a, wx[g], bv[g]);
    }
#pragma unroll
    for (int kt = 0; kt < 4; ++kt) {
      const v16h a = load_a_frag(&hA[0][0], H_, lane, kt);
#pragma unroll
      for (int g = 0; g < NGATE_; ++g) acc[g] = wmma_16x16x32(a, wh[g][kt], acc[g]);
    }
    __syncthreads();   // all waves done reading hA

    // --- wave-local cell update: gates straight from accumulators ------
    // acc[g] element r <-> (m = r + 8*hi, j = jcol); acc0..3 = i,f,g,o
#pragma unroll
    for (int r = 0; r < 8; ++r) {
      const float cn = fast_sigmoid(acc[1][r]) * c[r] +
                       fast_sigmoid(acc[0][r]) * fast_tanh(acc[2][r]);
      c[r] = cn;
      hA[r + 8 * hi][jcol] = (_Float16)(fast_sigmoid(acc[3][r]) * fast_tanh(cn));
    }
  }

  // final h_{T-1} tile
  __syncthreads();
  {
    const v8h* s = (const v8h*)(&hA[0][0]) + tid;
    v8h* d = (v8h*)(h1seq + ((size_t)blockIdx.x * T_ + (T_ - 1)) * (BT_ * H_)) + tid;
    *d = *s;
  }
}

// ---------------------------------------------------------------------------
// Phase B: layer-1 LSTM + FC.  gates = h1_t @ Wih1^T + h @ Whh1^T + bias
// 16 waves x 2 n-tiles; gates exchanged via LDS (keeps VGPRs < 256).
// ---------------------------------------------------------------------------
__global__ __launch_bounds__(NTHR_B_, 1)
void lstm_layer1_fc_kernel(const _Float16* __restrict__ h1seq,
                           const float* __restrict__ Wih,
                           const float* __restrict__ Whh,
                           const float* __restrict__ bih,
                           const float* __restrict__ bhh,
                           const float* __restrict__ Wfc,
                           const float* __restrict__ bfc,
                           float* __restrict__ out) {
  __shared__ alignas(16) _Float16 hA[BT_][H_];
  __shared__ float gates[BT_][G4H_];

  const int tid  = threadIdx.x;
  const int lane = tid & 31;
  const int w    = tid >> 5;
  const int hi   = (lane >> 4) & 1;
  const int b0   = blockIdx.x * BT_;

  v16h wi[NT_PW_][4], wh[NT_PW_][4];
  v8f  bv[NT_PW_];
#pragma unroll
  for (int nt = 0; nt < NT_PW_; ++nt) {
    const int n = w * (NT_PW_ * 16) + nt * 16 + (lane & 15);
    const float b = bih[n] + bhh[n];
#pragma unroll
    for (int r = 0; r < 8; ++r) bv[nt][r] = b;
#pragma unroll
    for (int kt = 0; kt < 4; ++kt) {
      wi[nt][kt] = load_b_frag_w(Wih, H_, n, lane, kt);
      wh[nt][kt] = load_b_frag_w(Whh, H_, n, lane, kt);
    }
  }
#pragma unroll
  for (int nt = 0; nt < NT_PW_; ++nt) {
    pin_acc(bv[nt]);
#pragma unroll
    for (int kt = 0; kt < 4; ++kt) { pin_frag(wi[nt][kt]); pin_frag(wh[nt][kt]); }
  }

  for (int i = tid; i < BT_ * H_; i += NTHR_B_) (&hA[0][0])[i] = (_Float16)0.0f;
  float c[4] = {0.0f, 0.0f, 0.0f, 0.0f};
  __syncthreads();

#pragma unroll 1
  for (int t = 0; t < T_; ++t) {
    const _Float16* h0 = h1seq + ((size_t)blockIdx.x * T_ + t) * (BT_ * H_);

    // warm WGP$ with the next h1 tile (4 KB across 512 lanes)
    if (t + 1 < T_) __builtin_prefetch(h0 + BT_ * H_ + tid * 4, 0, 0);

    v8f acc[NT_PW_];
    // 16 WMMAs per wave: input GEMM (A from global h1seq) + recurrent GEMM.
    // Bias splat enters as C of the first (kt=0) WMMA.
    {
      v16h a0 = load_a_frag(h0, H_, lane, 0);
      v16h a1 = load_a_frag(&hA[0][0], H_, lane, 0);
#pragma unroll
      for (int nt = 0; nt < NT_PW_; ++nt) {
        acc[nt] = wmma_16x16x32(a0, wi[nt][0], bv[nt]);
        acc[nt] = wmma_16x16x32(a1, wh[nt][0], acc[nt]);
      }
    }
#pragma unroll
    for (int kt = 1; kt < 4; ++kt) {
      v16h a0 = load_a_frag(h0, H_, lane, kt);
      v16h a1 = load_a_frag(&hA[0][0], H_, lane, kt);
#pragma unroll
      for (int nt = 0; nt < NT_PW_; ++nt) {
        acc[nt] = wmma_16x16x32(a0, wi[nt][kt], acc[nt]);
        acc[nt] = wmma_16x16x32(a1, wh[nt][kt], acc[nt]);
      }
    }
    // scatter C/D: element r -> (m = r + 8*hi, n = n0 + lane&15)
#pragma unroll
    for (int nt = 0; nt < NT_PW_; ++nt) {
      const int n = w * (NT_PW_ * 16) + nt * 16 + (lane & 15);
#pragma unroll
      for (int r = 0; r < 8; ++r) gates[r + 8 * hi][n] = acc[nt][r];
    }
    __syncthreads();   // gates visible; hA reads done

    // cell update: 4 cells per lane, branch-free
    {
      const int j  = w * 8 + (lane & 7);
      const int mb = (lane >> 3) * 4;
#pragma unroll
      for (int r = 0; r < 4; ++r) {
        const int m = mb + r;
        const float ig = gates[m][j];
        const float fg = gates[m][H_ + j];
        const float gg = gates[m][2 * H_ + j];
        const float og = gates[m][3 * H_ + j];
        const float cn = fast_sigmoid(fg) * c[r] + fast_sigmoid(ig) * fast_tanh(gg);
        c[r] = cn;
        hA[m][j] = (_Float16)(fast_sigmoid(og) * fast_tanh(cn));
      }
    }
    __syncthreads();   // h(t) visible for t+1
  }

  // FC: out[b] = h_T[b,:] . Wfc + bfc   (16 rows per block)
  if (tid < BT_) {
    float a = bfc[0];
#pragma unroll 8
    for (int j = 0; j < H_; ++j) a += (float)hA[tid][j] * Wfc[j];
    out[b0 + tid] = a;
  }
}

// ---------------------------------------------------------------------------
extern "C" void kernel_launch(void* const* d_in, const int* in_sizes, int n_in,
                              void* d_out, int out_size, void* d_ws, size_t ws_size,
                              hipStream_t stream) {
  const float* x    = (const float*)d_in[0];
  const float* Wih0 = (const float*)d_in[1];
  const float* Whh0 = (const float*)d_in[2];
  const float* bih0 = (const float*)d_in[3];
  const float* bhh0 = (const float*)d_in[4];
  const float* Wih1 = (const float*)d_in[5];
  const float* Whh1 = (const float*)d_in[6];
  const float* bih1 = (const float*)d_in[7];
  const float* bhh1 = (const float*)d_in[8];
  const float* Wfc  = (const float*)d_in[9];
  const float* bfc  = (const float*)d_in[10];

  // Workspace: layer-0 hidden sequence in f16, [B/16][T][16][128] tiles.
  // Requires B*T*H*2 = 128 MiB.
  _Float16* h1seq = (_Float16*)d_ws;
  (void)in_sizes; (void)n_in; (void)out_size; (void)ws_size;

  lstm_layer0_kernel<<<dim3(NBLK_), dim3(NTHR_A_), 0, stream>>>(
      x, Wih0, Whh0, bih0, bhh0, h1seq);
  lstm_layer1_fc_kernel<<<dim3(NBLK_), dim3(NTHR_B_), 0, stream>>>(
      h1seq, Wih1, Whh1, bih1, bhh1, Wfc, bfc, (float*)d_out);
}
